// STFT_42923903156611
// MI455X (gfx1250) — matrix operand order repaired and checked by reading, not compiled, not run on hardware
//
#include <hip/hip_runtime.h>
#include <hip/hip_bf16.h>

// ---------------------------------------------------------------------------
// STFT as split-precision f16 WMMA GEMM for gfx1250 (MI455X).
//
//   Re[b,k,t] =  sum_n [cos(2*pi*k*n/N)*w[n]] * x[t*HOP + n - 384]
//   Im[b,k,t] =  sum_n [-sin(2*pi*k*n/N)*w[n]] * x[t*HOP + n - 384]
//
// The window is folded into the twiddle tables (both depend only on k,n),
// built in double precision with exact integer phase reduction.  The signal
// is pre-split once per launch into zero-padded f16 hi/lo planes so the main
// loop is pure {global_load_b128 -> v_wmma_f32_16x16x32_f16} with no VALU
// conversion work and no bounds checks.  Split-precision GEMM:
//   D = Ah*Bh + Ah*Bl + Al*Bh   (f32 accumulate; dropped Al*Bl ~ 2^-22 rel)
// ---------------------------------------------------------------------------

#define N_FFT   1024
#define HOP     256
#define SLEN    524288
#define FRAMES  2048        // (SLEN + 768 - 1024)/256 + 1
#define BINS    513
#define KTILES  33          // 528 rows >= 513 bins
#define BATCH   16
#define PADL    384         // left zero-pad ((1024-256)/2)
#define PADROW  525312      // 384 + 524288 + 640, rounded up (mult of 256)
#define TWT     (KTILES * 16 * N_FFT)           // halves per twiddle table

typedef __attribute__((ext_vector_type(16))) _Float16 v16h;
typedef __attribute__((ext_vector_type(8)))  _Float16 v8h;
typedef __attribute__((ext_vector_type(8)))  float    v8f;

union Frag16 { v16h v; v8h h[2]; _Float16 e[16]; };

// ---------------------------------------------------------------------------
// Setup 1: window-folded cos/sin hi/lo twiddle tables in d_ws.
// tab[0..T) = cos_hi, [T..2T) = cos_lo, [2T..3T) = (-sin)_hi, [3T..4T) = lo.
// Row r = frequency bin k (row-major over n).  4*T halves = 4.3 MB -> L2.
// ---------------------------------------------------------------------------
__global__ void stft_twiddle_setup(const float* __restrict__ win,
                                   _Float16* __restrict__ tab) {
    int e = blockIdx.x * blockDim.x + threadIdx.x;
    if (e >= TWT) return;
    int k = e >> 10;
    int n = e & (N_FFT - 1);
    int phase = (k * n) & (N_FFT - 1);          // exact: k*n < 2^24
    double ang = -6.283185307179586476925286766559 * (double)phase / 1024.0;
    double wv  = (double)win[n];
    float c  = (float)(cos(ang) * wv);          //  cos(2*pi*k*n/N) * w[n]
    float sn = (float)(sin(ang) * wv);          // -sin(2*pi*k*n/N) * w[n]
    _Float16 ch = (_Float16)c;
    _Float16 cl = (_Float16)(c - (float)ch);
    _Float16 sh = (_Float16)sn;
    _Float16 sl = (_Float16)(sn - (float)sh);
    tab[e]           = ch;
    tab[TWT + e]     = cl;
    tab[2 * TWT + e] = sh;
    tab[3 * TWT + e] = sl;
}

// ---------------------------------------------------------------------------
// Setup 2: zero-padded f16 hi/lo split of the signal (per batch row).
// xh/xl are (16, PADROW); position p maps to source sample p - 384.
// ---------------------------------------------------------------------------
__global__ void stft_xsplit(const float* __restrict__ x,
                            _Float16* __restrict__ xh,
                            _Float16* __restrict__ xl) {
    long long e = (long long)blockIdx.x * blockDim.x + threadIdx.x;
    if (e >= (long long)BATCH * PADROW) return;
    int b = (int)(e / PADROW);
    int p = (int)(e % PADROW);
    long long si = (long long)p - PADL;
    float v = (si >= 0 && si < (long long)SLEN)
                  ? x[(long long)b * SLEN + si] : 0.0f;
    _Float16 h = (_Float16)v;
    xh[e] = h;
    xl[e] = (_Float16)(v - (float)h);
}

// ---------------------------------------------------------------------------
// Main kernel.  Block = 256 threads = 8 wave32.  Each wave computes TWO
// 16(bins) x 16(frames) tiles (re+im), sharing the A fragments: per K-chunk
// 8 A loads + 8 B loads (all b128) feed 12 v_wmma_f32_16x16x32_f16.
// Grid = (FRAMES/256, KTILES, BATCH) = (8, 33, 16).
//
// PRESPLIT=true : B from padded f16 hi/lo planes (no guards, no VALU).
// PRESPLIT=false: fallback, inline split from f32 x with branchless clamp.
// ---------------------------------------------------------------------------
template <bool PRESPLIT>
__global__ __launch_bounds__(256) void stft_wmma_kernel(
    const float* __restrict__ x,
    const _Float16* __restrict__ tab,
    const _Float16* __restrict__ xh,
    const _Float16* __restrict__ xl,
    float* __restrict__ out) {
    const int lane  = threadIdx.x & 31;
    const int wave  = threadIdx.x >> 5;
    const int m     = lane & 15;     // A-row / B-column / D-column sub-index
    const int hb    = lane >> 4;     // lane half (selects K sub-range)
    const int ttile = blockIdx.x;
    const int ktile = blockIdx.y;
    const int b     = blockIdx.z;

    const int t0 = ttile * 256 + wave * 16 + m;   // frame of tile 0
    const int t1 = t0 + 128;                      // frame of tile 1
    const int rowbase = (ktile * 16 + m) * N_FFT;
    const int K0 = hb * 16;                       // this lane's first B row

    const _Float16* tc_h = tab;
    const _Float16* tc_l = tab + TWT;
    const _Float16* ts_h = tab + 2 * TWT;
    const _Float16* ts_l = tab + 3 * TWT;

    // B base pointers (presplit) or sample bases (fallback)
    const _Float16 *ph0, *pl0, *ph1, *pl1;
    const float* xb = nullptr;
    long long sb0 = 0, sb1 = 0;
    bool interior = true;
    if (PRESPLIT) {
        const _Float16* xhb = xh + (long long)b * PADROW;
        const _Float16* xlb = xl + (long long)b * PADROW;
        ph0 = xhb + t0 * HOP + K0;   pl0 = xlb + t0 * HOP + K0;
        ph1 = xhb + t1 * HOP + K0;   pl1 = xlb + t1 * HOP + K0;
    } else {
        xb  = x + (long long)b * SLEN;
        sb0 = (long long)t0 * HOP - PADL + K0;
        sb1 = (long long)t1 * HOP - PADL + K0;
        interior = (ttile != 0) && (ttile != (int)gridDim.x - 1);
    }

    v8f re0 = {}, im0 = {}, re1 = {}, im1 = {};

    for (int kc = 0; kc < N_FFT / 32; ++kc) {
        // ---- B fragments (32x16 f16): lanes 0-15 K=0..15, 16-31 K=16..31
        Frag16 b0h, b0l, b1h, b1l;
        if (PRESPLIT) {
            b0h.v = *(const v16h*)(ph0 + kc * 32);   // 32B aligned -> 2x b128
            b0l.v = *(const v16h*)(pl0 + kc * 32);
            b1h.v = *(const v16h*)(ph1 + kc * 32);
            b1l.v = *(const v16h*)(pl1 + kc * 32);
        } else {
            float s0[16], s1[16];
            if (interior) {
                const float4* p0 = (const float4*)(xb + sb0 + kc * 32);
                const float4* p1 = (const float4*)(xb + sb1 + kc * 32);
#pragma unroll
                for (int q = 0; q < 4; ++q) {
                    float4 a = p0[q], c = p1[q];
                    s0[4 * q] = a.x; s0[4 * q + 1] = a.y;
                    s0[4 * q + 2] = a.z; s0[4 * q + 3] = a.w;
                    s1[4 * q] = c.x; s1[4 * q + 1] = c.y;
                    s1[4 * q + 2] = c.z; s1[4 * q + 3] = c.w;
                }
            } else {
#pragma unroll
                for (int j = 0; j < 16; ++j) {       // branchless clamp+mask
                    long long si = sb0 + kc * 32 + j;
                    long long sc = si < 0 ? 0 : (si > SLEN - 1 ? SLEN - 1 : si);
                    float v = xb[sc];
                    s0[j] = (si == sc) ? v : 0.0f;
                    si = sb1 + kc * 32 + j;
                    sc = si < 0 ? 0 : (si > SLEN - 1 ? SLEN - 1 : si);
                    v = xb[sc];
                    s1[j] = (si == sc) ? v : 0.0f;
                }
            }
#pragma unroll
            for (int j = 0; j < 16; ++j) {
                _Float16 h = (_Float16)s0[j];
                b0h.e[j] = h; b0l.e[j] = (_Float16)(s0[j] - (float)h);
                h = (_Float16)s1[j];
                b1h.e[j] = h; b1l.e[j] = (_Float16)(s1[j] - (float)h);
            }
        }

        // ---- A fragments (16x32 f16): lanes 0-15 K=0..15,
        //      lanes 16-31 K=16..31; shared by both frame tiles.
        const int g0 = kc * 32 + hb * 16;
        const int g1 = g0 + 8;
        Frag16 ach, acl, ash, asl;
        ach.h[0] = *(const v8h*)(tc_h + rowbase + g0);
        ach.h[1] = *(const v8h*)(tc_h + rowbase + g1);
        acl.h[0] = *(const v8h*)(tc_l + rowbase + g0);
        acl.h[1] = *(const v8h*)(tc_l + rowbase + g1);
        ash.h[0] = *(const v8h*)(ts_h + rowbase + g0);
        ash.h[1] = *(const v8h*)(ts_h + rowbase + g1);
        asl.h[0] = *(const v8h*)(ts_l + rowbase + g0);
        asl.h[1] = *(const v8h*)(ts_l + rowbase + g1);

        // ---- 12 WMMAs: D = Ah*Bh + Ah*Bl + Al*Bh for re/im x 2 tiles
        re0 = __builtin_amdgcn_wmma_f32_16x16x32_f16(
            false, ach.v, false, b0h.v, (short)0, re0, false, false);
        re1 = __builtin_amdgcn_wmma_f32_16x16x32_f16(
            false, ach.v, false, b1h.v, (short)0, re1, false, false);
        im0 = __builtin_amdgcn_wmma_f32_16x16x32_f16(
            false, ash.v, false, b0h.v, (short)0, im0, false, false);
        im1 = __builtin_amdgcn_wmma_f32_16x16x32_f16(
            false, ash.v, false, b1h.v, (short)0, im1, false, false);
        re0 = __builtin_amdgcn_wmma_f32_16x16x32_f16(
            false, ach.v, false, b0l.v, (short)0, re0, false, false);
        re1 = __builtin_amdgcn_wmma_f32_16x16x32_f16(
            false, ach.v, false, b1l.v, (short)0, re1, false, false);
        im0 = __builtin_amdgcn_wmma_f32_16x16x32_f16(
            false, ash.v, false, b0l.v, (short)0, im0, false, false);
        im1 = __builtin_amdgcn_wmma_f32_16x16x32_f16(
            false, ash.v, false, b1l.v, (short)0, im1, false, false);
        re0 = __builtin_amdgcn_wmma_f32_16x16x32_f16(
            false, acl.v, false, b0h.v, (short)0, re0, false, false);
        re1 = __builtin_amdgcn_wmma_f32_16x16x32_f16(
            false, acl.v, false, b1h.v, (short)0, re1, false, false);
        im0 = __builtin_amdgcn_wmma_f32_16x16x32_f16(
            false, asl.v, false, b0h.v, (short)0, im0, false, false);
        im1 = __builtin_amdgcn_wmma_f32_16x16x32_f16(
            false, asl.v, false, b1h.v, (short)0, im1, false, false);
    }

    // ---- store D: lane = column N (frame), VGPR v = row M = v + 8*hb.
    float* out_re = out;
    float* out_im = out + (long long)BATCH * BINS * FRAMES;
    const long long base = (long long)b * BINS;
#pragma unroll
    for (int v = 0; v < 8; ++v) {
        int kk = ktile * 16 + v + 8 * hb;
        if (kk < BINS) {
            long long o0 = (base + kk) * FRAMES + t0;
            out_re[o0] = re0[v];
            out_im[o0] = im0[v];
            out_re[o0 + 128] = re1[v];
            out_im[o0 + 128] = im1[v];
        }
    }
}

// ---------------------------------------------------------------------------
// kernel_launch.  d_ws layout (halves): [0,4T) twiddles; [4T, +16*PADROW) xh;
// then xl.  Needs 37,945,344 bytes for the fast path; falls back to inline
// split (still twiddle-table based, 4.3 MB) if d_ws is smaller.  The choice
// depends only on ws_size (fixed per run) -> deterministic.
// ---------------------------------------------------------------------------
extern "C" void kernel_launch(void* const* d_in, const int* in_sizes, int n_in,
                              void* d_out, int out_size, void* d_ws, size_t ws_size,
                              hipStream_t stream) {
    (void)in_sizes; (void)n_in; (void)out_size;
    const float* x   = (const float*)d_in[0];
    const float* win = (const float*)d_in[1];
    _Float16* tab = (_Float16*)d_ws;
    _Float16* xh  = tab + 4 * (size_t)TWT;
    _Float16* xl  = xh + (size_t)BATCH * PADROW;

    const size_t need_tw   = (size_t)4 * TWT * 2;
    const size_t need_full = need_tw + (size_t)2 * BATCH * PADROW * 2;

    stft_twiddle_setup<<<(TWT + 255) / 256, 256, 0, stream>>>(win, tab);

    dim3 grid(FRAMES / 256, KTILES, BATCH);   // (8, 33, 16)
    if (ws_size >= need_full) {
        long long nsplit = (long long)BATCH * PADROW;
        stft_xsplit<<<(unsigned)((nsplit + 255) / 256), 256, 0, stream>>>(
            x, xh, xl);
        stft_wmma_kernel<true><<<grid, 256, 0, stream>>>(
            x, tab, xh, xl, (float*)d_out);
    } else {
        stft_wmma_kernel<false><<<grid, 256, 0, stream>>>(
            x, tab, xh, xl, (float*)d_out);
    }
}
